// Critic_12635793785256
// MI455X (gfx1250) — matrix operand order repaired and checked by reading, not compiled, not run on hardware
//
#include <hip/hip_runtime.h>

typedef _Float16 v16h __attribute__((ext_vector_type(16)));
typedef _Float16 v8h  __attribute__((ext_vector_type(8)));
typedef _Float16 v4h  __attribute__((ext_vector_type(4)));
typedef float    v8f  __attribute__((ext_vector_type(8)));

#define EMBED      64
#define EDGE_IN    16
#define HIDDEN     256
#define CONCAT     272     // 4*EMBED + EDGE_IN
#define KPAD       288     // 9 * 32, zero-padded K for WMMA k-steps
#define KSTEPS     9
#define LDS_STRIDE 296     // halves per LDS row (592B = 148 dwords) -> bank-conflict-free A frags
#define CHUNKS     (LDS_STRIDE / 4)   // 74 float4-granular chunks per row
#define TILE_E     64      // edges per workgroup (4 x 16-row M tiles)
#define NT         256     // threads per block (8 waves)

// ---------------------------------------------------------------------------
// One-time pack: W1 [272 x 256] row-major f32  ->  w1h [col][KPAD] f16,
// zero-padded K 272..287. Each lane's WMMA B-fragment becomes two
// contiguous 16B global loads.
// ---------------------------------------------------------------------------
__global__ void pack_w1_kernel(const float* __restrict__ W1,
                               _Float16* __restrict__ w1h) {
    int idx = blockIdx.x * blockDim.x + threadIdx.x;   // idx = c*KPAD + k
    if (idx >= HIDDEN * KPAD) return;
    int c = idx / KPAD;
    int k = idx - c * KPAD;
    float v = (k < CONCAT) ? W1[k * HIDDEN + c] : 0.0f;
    w1h[idx] = (_Float16)v;
}

// ---------------------------------------------------------------------------
// Fused: gather+concat -> f16 LDS tile -> WMMA GEMM (64x288 @ 288x256)
// -> bias/ReLU/W2 dot -> 16-lane butterfly reduce -> q per edge.
// Block = 256 threads (8 waves). Wave owns 2 column tiles; each B fragment
// feeds 4 WMMAs (4 edge sub-tiles) before being replaced.
// ---------------------------------------------------------------------------
__global__ __launch_bounds__(NT) void edge_mlp_kernel(
    const float*     __restrict__ node_emb,    // [N, 64]
    const float*     __restrict__ global_ctx,  // [G, 128]
    const float*     __restrict__ edge_attr,   // [E, 16]
    const _Float16*  __restrict__ w1h,         // [256, KPAD] packed f16
    const float*     __restrict__ b1,          // [256]
    const float*     __restrict__ W2,          // [256]
    const float*     __restrict__ b2,          // [1]
    const long long* __restrict__ edge_index,  // [2, E] int64
    const long long* __restrict__ batch,       // [N] int64
    float*           __restrict__ out,         // [E]
    int E)
{
    __shared__ __align__(16) _Float16 xtile[TILE_E][LDS_STRIDE];
    __shared__ float qacc[TILE_E];
    __shared__ int   sidx[TILE_E], didx[TILE_E], gidx[TILE_E];

    const int tid  = threadIdx.x;
    const int lane = tid & 31;
    const int wave = tid >> 5;
    const long long ebase = (long long)blockIdx.x * TILE_E;

    // Preload edge indices (int64 -> int) and init q accumulators.
    if (tid < TILE_E) {
        long long e = ebase + tid;
        int s = 0, d = 0, g = 0;
        if (e < (long long)E) {
            s = (int)edge_index[e];
            d = (int)edge_index[(long long)E + e];
            g = (int)batch[s];
        }
        sidx[tid] = s; didx[tid] = d; gidx[tid] = g;
        qacc[tid] = 0.0f;
    }
    __syncthreads();

    // Build the padded f16 feature tile in float4-granular chunks.
    // Row layout: [0,64) src emb | [64,128) dst emb | [128,144) attr |
    //             [144,272) ctx | [272,296) zeros. All boundaries % 4 == 0.
    for (int idx = tid; idx < TILE_E * CHUNKS; idx += NT) {
        int r = idx / CHUNKS;
        int k = (idx - r * CHUNKS) * 4;
        long long e = ebase + r;
        float4 v = make_float4(0.0f, 0.0f, 0.0f, 0.0f);
        if (e < (long long)E && k < CONCAT) {
            const float* p;
            if (k < EMBED) {
                p = node_emb + (long long)sidx[r] * EMBED + k;
            } else if (k < 2 * EMBED) {
                p = node_emb + (long long)didx[r] * EMBED + (k - EMBED);
            } else if (k < 2 * EMBED + EDGE_IN) {
                p = edge_attr + e * EDGE_IN + (k - 2 * EMBED);
            } else {
                p = global_ctx + (long long)gidx[r] * (2 * EMBED) + (k - (2 * EMBED + EDGE_IN));
            }
            v = *(const float4*)p;
        }
        v4h h;
        h[0] = (_Float16)v.x; h[1] = (_Float16)v.y;
        h[2] = (_Float16)v.z; h[3] = (_Float16)v.w;
        *(v4h*)(&xtile[r][k]) = h;
    }
    __syncthreads();

    const int half = lane >> 4;       // 0: lanes 0-15, 1: lanes 16-31
    const int l16  = lane & 15;

    for (int ctp = 0; ctp < 2; ++ctp) {
        const int ct  = wave * 2 + ctp;     // column tile 0..15
        const int col = ct * 16 + l16;      // this lane's H column

        v8f acc[4] = {v8f{}, v8f{}, v8f{}, v8f{}};

        // B fragment base: lane<16 -> col, K k0..k0+7,k0+16..k0+23 ; lane>=16 -> K k0+8..k0+15,k0+24..k0+31
        const _Float16* bbase = w1h + (size_t)col * KPAD + half * 8;

        for (int ks = 0; ks < KSTEPS; ++ks) {
            const int k0 = ks * 32;

            union { v16h v; v8h h[2]; } bfrag;
            bfrag.h[0] = *(const v8h*)(bbase + k0);
            bfrag.h[1] = *(const v8h*)(bbase + k0 + 16);

            // A fragment (16-bit A 16x32 layout):
            //   lane<16 : v0-3 = K k0..k0+7,  v4-7 = K k0+16..k0+23
            //   lane>=16: v0-3 = K k0+8..+15, v4-7 = K k0+24..+31
            const int aoff = k0 + half * 8;
            #pragma unroll
            for (int et = 0; et < 4; ++et) {
                union { v16h v; v8h h[2]; } afrag;
                afrag.h[0] = *(const v8h*)(&xtile[et * 16 + l16][aoff]);
                afrag.h[1] = *(const v8h*)(&xtile[et * 16 + l16][aoff + 16]);
                acc[et] = __builtin_amdgcn_wmma_f32_16x16x32_f16(
                    false, afrag.v, false, bfrag.v, (short)0, acc[et], false, false);
            }
        }

        // Epilogue: h = relu(acc + b1[col]); partial q = h * W2[col], then
        // reduce across the 16 lanes sharing each output row.
        // (C/D layout: VGPR i -> M = i + 8*half, N = l16; xor masks 1,2,4,8
        //  never cross the 16-lane half boundary.)
        const float bias = b1[col];
        const float w2v  = W2[col];
        #pragma unroll
        for (int et = 0; et < 4; ++et) {
            float s[8];
            #pragma unroll
            for (int i = 0; i < 8; ++i) {
                float h = acc[et][i] + bias;
                h = h > 0.0f ? h : 0.0f;
                s[i] = h * w2v;
            }
            #pragma unroll
            for (int m = 1; m < 16; m <<= 1) {
                #pragma unroll
                for (int i = 0; i < 8; ++i) {
                    s[i] += __shfl_xor(s[i], m, 32);
                }
            }
            if (l16 == 0) {
                const int mb = et * 16 + half * 8;
                #pragma unroll
                for (int i = 0; i < 8; ++i) {
                    atomicAdd(&qacc[mb + i], s[i]);
                }
            }
        }
    }
    __syncthreads();

    if (tid < TILE_E) {
        long long e = ebase + tid;
        if (e < (long long)E) out[e] = qacc[tid] + b2[0];
    }
}

// ---------------------------------------------------------------------------
// Launch wrapper
// ---------------------------------------------------------------------------
extern "C" void kernel_launch(void* const* d_in, const int* in_sizes, int n_in,
                              void* d_out, int out_size, void* d_ws, size_t ws_size,
                              hipStream_t stream) {
    const float*     node_emb   = (const float*)d_in[0];
    const float*     global_ctx = (const float*)d_in[1];
    const float*     edge_attr  = (const float*)d_in[2];
    const float*     W1         = (const float*)d_in[3];
    const float*     b1         = (const float*)d_in[4];
    const float*     W2         = (const float*)d_in[5];
    const float*     b2         = (const float*)d_in[6];
    const long long* edge_index = (const long long*)d_in[7];
    const long long* batch      = (const long long*)d_in[8];
    float*           out        = (float*)d_out;

    const int E = in_sizes[7] / 2;

    _Float16* w1h = (_Float16*)d_ws;   // 256*288*2 = 147,456 bytes of scratch

    const int packN = HIDDEN * KPAD;
    pack_w1_kernel<<<(packN + 255) / 256, 256, 0, stream>>>(W1, w1h);

    const int nblocks = (E + TILE_E - 1) / TILE_E;
    edge_mlp_kernel<<<nblocks, NT, 0, stream>>>(
        node_emb, global_ctx, edge_attr, w1h, b1, W2, b2,
        edge_index, batch, out, E);
}
